// PachifiedEmbed_5987184411247
// MI455X (gfx1250) — matrix operand order repaired and checked by reading, not compiled, not run on hardware
//
#include <hip/hip_runtime.h>
#include <math.h>

// Problem constants (match reference)
#define B_  2
#define NPTS_ 32768
#define T_  32768
#define P_  512
#define C_  4
#define D_  768
#define TWO_D_ 1536
#define ASTR 776           // padded LDS row stride (elements); 776*2=1552 bytes, 16B aligned
#define MR  32             // token rows per block (two 16-row WMMA fragments)
#define NW 8               // waves per block
#define TPB 256

typedef __attribute__((ext_vector_type(16))) __bf16 v16bf;
typedef __attribute__((ext_vector_type(8)))  float  v8f;

union FragBF { v16bf v; uint4 q[2]; };

__device__ __forceinline__ unsigned short f2bf(float f) {
    unsigned u = __float_as_uint(f);
    u += 0x7fffu + ((u >> 16) & 1u);       // round-to-nearest-even
    return (unsigned short)(u >> 16);
}
__device__ __forceinline__ float bf2f(unsigned short h) {
    return __uint_as_float(((unsigned)h) << 16);
}
__device__ __forceinline__ float gelu_exact(float x) {
    return 0.5f * x * (1.0f + erff(x * 0.70710678118654752f));
}

// A-fragment (16x32 bf16, MxK): lane ln -> row m; half selects K-striping.
// elements 0..7 -> K = kk + 8*half + i ; elements 8..15 -> K = kk + 16 + 8*half + i
__device__ __forceinline__ v16bf load_a_frag(const unsigned short* Abuf, int stride,
                                             int m, int half, int kk) {
    FragBF f;
    const unsigned short* p = Abuf + m * stride + kk + 8 * half;
    f.q[0] = *(const uint4*)(p);
    f.q[1] = *(const uint4*)(p + 16);
    return f.v;
}

// B-fragment (32x16 bf16, KxN): lane ln = column n; elements 0..7 -> K = kk + 8*half + i ; elements 8..15 -> K = kk + 16 + 8*half + i
// Weights are stored transposed-bf16: wT[n][k], row length Kdim (contiguous K).
__device__ __forceinline__ v16bf load_b_frag(const unsigned short* __restrict__ wT,
                                             long long n, int Kdim, int half, int kk) {
    FragBF f;
    const unsigned short* p = wT + n * (long long)Kdim + kk + 8 * half;
    f.q[0] = *(const uint4*)(p);
    f.q[1] = *(const uint4*)(p + 16);
    return f.v;
}

// ---------------- weight transpose + bf16 convert: dst[n*K + k] = bf16(src[k*N + n])
__global__ void convT_kernel(const float* __restrict__ src, unsigned short* __restrict__ dst,
                             int K, int N) {
    long long idx = (long long)blockIdx.x * blockDim.x + threadIdx.x;
    if (idx >= (long long)K * N) return;
    int n = (int)(idx / K);
    int k = (int)(idx % K);
    dst[idx] = f2bf(src[(long long)k * N + n]);
}

// ---------------- seg[t] = searchsorted(patch_offset[1:], t, 'right')
__global__ void seg_kernel(const int* __restrict__ poff, int* __restrict__ seg) {
    int t = blockIdx.x * blockDim.x + threadIdx.x;
    if (t >= T_) return;
    int lo = 0, hi = P_;
    while (lo < hi) {
        int mid = (lo + hi) >> 1;
        if (poff[mid + 1] > t) hi = mid; else lo = mid + 1;
    }
    seg[t] = lo;
}

// ---------------- Stage 1: h1 = gelu(ln(x@w1a+b1a)) @ w1b + b1b  (bf16 out)
__global__ void __launch_bounds__(TPB)
stage1_kernel(const float* __restrict__ latent, const int* __restrict__ patch_index,
              const float* __restrict__ w1a, const float* __restrict__ b1a,
              const unsigned short* __restrict__ w1bT, const float* __restrict__ b1b,
              unsigned short* __restrict__ h1) {
    __shared__ __align__(16) unsigned short Abuf[MR * ASTR];
    __shared__ float xsh[MR][4];
    __shared__ float redS[MR][8];
    __shared__ float redQ[MR][8];
    __shared__ float stats[MR][2];

    const int tid = threadIdx.x;
    const long long r0 = (long long)blockIdx.x * MR;   // global row (b*T + t)

    if (tid < MR * 4) {
        int m = tid >> 2, c = tid & 3;
        long long row = r0 + m;
        int t = (int)(row % T_);
        int b = (int)(row / T_);
        int pi = patch_index[t];
        xsh[m][c] = latent[((long long)b * NPTS_ + pi) * C_ + c];
    }
    __syncthreads();

    // pass 1: LN stats over hpre = x@w1a + b1a (recompute, no big buffer). 8 threads/row.
    {
        const int m = tid >> 3, cg = tid & 7;
        const float x0 = xsh[m][0], x1 = xsh[m][1], x2 = xsh[m][2], x3 = xsh[m][3];
        float s = 0.f, q = 0.f;
        for (int d = cg; d < D_; d += 8) {
            float v = b1a[d] + x0 * w1a[d] + x1 * w1a[D_ + d]
                             + x2 * w1a[2 * D_ + d] + x3 * w1a[3 * D_ + d];
            s += v; q += v * v;
        }
        redS[m][cg] = s; redQ[m][cg] = q;
    }
    __syncthreads();
    if (tid < MR) {
        float s = 0.f, q = 0.f;
        for (int i = 0; i < 8; ++i) { s += redS[tid][i]; q += redQ[tid][i]; }
        float mu  = s * (1.0f / D_);
        float var = q * (1.0f / D_) - mu * mu;
        stats[tid][0] = mu;
        stats[tid][1] = rsqrtf(var + 1e-6f);
    }
    __syncthreads();

    // pass 2: recompute, LN, gelu -> bf16 A tile
    {
        const int m = tid >> 3, cg = tid & 7;
        const float x0 = xsh[m][0], x1 = xsh[m][1], x2 = xsh[m][2], x3 = xsh[m][3];
        const float mu = stats[m][0], rstd = stats[m][1];
        for (int d = cg; d < D_; d += 8) {
            float v = b1a[d] + x0 * w1a[d] + x1 * w1a[D_ + d]
                             + x2 * w1a[2 * D_ + d] + x3 * w1a[3 * D_ + d];
            v = (v - mu) * rstd;
            Abuf[m * ASTR + d] = f2bf(gelu_exact(v));
        }
    }
    __syncthreads();

    // WMMA GEMM: (32 x 768) += A(32x768) * w1b(768x768); two row-fragments share each B
    const int wv = tid >> 5, lane = tid & 31, half = lane >> 4, ln = lane & 15;
    v8f acc[2][6];
    #pragma unroll
    for (int rb = 0; rb < 2; ++rb)
        #pragma unroll
        for (int j = 0; j < 6; ++j)
            #pragma unroll
            for (int e = 0; e < 8; ++e) acc[rb][j][e] = 0.f;

    for (int kk = 0; kk < D_; kk += 32) {
        v16bf a0 = load_a_frag(Abuf, ASTR, ln,      half, kk);
        v16bf a1 = load_a_frag(Abuf, ASTR, 16 + ln, half, kk);
        #pragma unroll
        for (int j = 0; j < 6; ++j) {
            int n = wv * 96 + j * 16 + ln;
            v16bf bb = load_b_frag(w1bT, n, D_, half, kk);
            acc[0][j] = __builtin_amdgcn_wmma_f32_16x16x32_bf16(
                false, a0, false, bb, (short)0, acc[0][j], false, false);
            acc[1][j] = __builtin_amdgcn_wmma_f32_16x16x32_bf16(
                false, a1, false, bb, (short)0, acc[1][j], false, false);
        }
    }
    // store h1 (bf16) with bias. C layout: row m = rb*16 + v + 8*half, col n = base + ln
    #pragma unroll
    for (int j = 0; j < 6; ++j) {
        int n = wv * 96 + j * 16 + ln;
        float bias = b1b[n];
        #pragma unroll
        for (int rb = 0; rb < 2; ++rb)
            #pragma unroll
            for (int v = 0; v < 8; ++v) {
                int m = rb * 16 + v + 8 * half;
                h1[(r0 + m) * D_ + n] = f2bf(acc[rb][j][v] + bias);
            }
    }
}

// ---------------- segment max of bf16 h1 -> y (fp32)
__global__ void segmax1_kernel(const unsigned short* __restrict__ h1,
                               const int* __restrict__ poff, float* __restrict__ y) {
    const int p = blockIdx.x, b = blockIdx.y;
    const int t0 = poff[p], t1 = poff[p + 1];
    for (int d = threadIdx.x; d < D_; d += blockDim.x) {
        float m = -INFINITY;
        for (int t = t0; t < t1; ++t)
            m = fmaxf(m, bf2f(h1[((long long)b * T_ + t) * D_ + d]));
        y[((long long)b * P_ + p) * D_ + d] = m;
    }
}

// ---------------- Stage 2: raw = gelu(ln(concat(y[seg],h1)@w2a+b2a)*g2+beta2) @ w2b + b2b
__global__ void __launch_bounds__(TPB)
stage2_kernel(const unsigned short* __restrict__ h1, const float* __restrict__ y,
              const int* __restrict__ seg,
              const unsigned short* __restrict__ w2aT, const float* __restrict__ b2a,
              const float* __restrict__ g2, const float* __restrict__ beta2,
              const unsigned short* __restrict__ w2bT, const float* __restrict__ b2b,
              float* __restrict__ raw_out) {
    __shared__ __align__(16) unsigned short Abuf[MR * ASTR];
    __shared__ int   segsh[MR];
    __shared__ float redS[MR][NW];
    __shared__ float redQ[MR][NW];
    __shared__ float stats[MR][2];

    const int tid = threadIdx.x;
    const long long r0 = (long long)blockIdx.x * MR;
    const int b = (int)(r0 / T_);

    if (tid < MR) segsh[tid] = seg[(int)((r0 + tid) % T_)];
    __syncthreads();

    const int wv = tid >> 5, lane = tid & 31, half = lane >> 4, ln = lane & 15;
    v8f acc[2][6];
    #pragma unroll
    for (int rb = 0; rb < 2; ++rb)
        #pragma unroll
        for (int j = 0; j < 6; ++j)
            #pragma unroll
            for (int e = 0; e < 8; ++e) acc[rb][j][e] = 0.f;

    // ---- phase Y: K half 0..767 from pooled y[b, seg[t], :]
    for (int idx = tid; idx < MR * D_; idx += TPB) {
        int m = idx / D_, k = idx % D_;
        Abuf[m * ASTR + k] = f2bf(y[((long long)b * P_ + segsh[m]) * D_ + k]);
    }
    __syncthreads();
    for (int kk = 0; kk < D_; kk += 32) {
        v16bf a0 = load_a_frag(Abuf, ASTR, ln,      half, kk);
        v16bf a1 = load_a_frag(Abuf, ASTR, 16 + ln, half, kk);
        #pragma unroll
        for (int j = 0; j < 6; ++j) {
            int n = wv * 96 + j * 16 + ln;
            v16bf bb = load_b_frag(w2aT, n, TWO_D_, half, kk);
            acc[0][j] = __builtin_amdgcn_wmma_f32_16x16x32_bf16(
                false, a0, false, bb, (short)0, acc[0][j], false, false);
            acc[1][j] = __builtin_amdgcn_wmma_f32_16x16x32_bf16(
                false, a1, false, bb, (short)0, acc[1][j], false, false);
        }
    }
    __syncthreads();
    // ---- phase H: K half 768..1535 from h1
    for (int idx = tid; idx < MR * D_; idx += TPB) {
        int m = idx / D_, k = idx % D_;
        Abuf[m * ASTR + k] = h1[(r0 + m) * D_ + k];
    }
    __syncthreads();
    for (int kk = 0; kk < D_; kk += 32) {
        v16bf a0 = load_a_frag(Abuf, ASTR, ln,      half, kk);
        v16bf a1 = load_a_frag(Abuf, ASTR, 16 + ln, half, kk);
        #pragma unroll
        for (int j = 0; j < 6; ++j) {
            int n = wv * 96 + j * 16 + ln;
            v16bf bb = load_b_frag(w2aT, n, TWO_D_, half, kk + D_);
            acc[0][j] = __builtin_amdgcn_wmma_f32_16x16x32_bf16(
                false, a0, false, bb, (short)0, acc[0][j], false, false);
            acc[1][j] = __builtin_amdgcn_wmma_f32_16x16x32_bf16(
                false, a1, false, bb, (short)0, acc[1][j], false, false);
        }
    }

    // ---- LN stats straight from accumulators (rows m = rb*16 + v + 8*half)
    float biasA[6];
    #pragma unroll
    for (int j = 0; j < 6; ++j) biasA[j] = b2a[wv * 96 + j * 16 + ln];

    float s[2][8], q[2][8];
    #pragma unroll
    for (int rb = 0; rb < 2; ++rb)
        #pragma unroll
        for (int v = 0; v < 8; ++v) {
            float ss = 0.f, qq = 0.f;
            #pragma unroll
            for (int j = 0; j < 6; ++j) {
                float x = acc[rb][j][v] + biasA[j];
                ss += x; qq += x * x;
            }
            s[rb][v] = ss; q[rb][v] = qq;
        }
    // butterfly within each 16-lane group (covers this wave's 96 columns)
    #pragma unroll
    for (int off = 8; off >= 1; off >>= 1) {
        #pragma unroll
        for (int rb = 0; rb < 2; ++rb)
            #pragma unroll
            for (int v = 0; v < 8; ++v) {
                s[rb][v] += __shfl_xor(s[rb][v], off, 16);
                q[rb][v] += __shfl_xor(q[rb][v], off, 16);
            }
    }
    if (ln == 0) {
        #pragma unroll
        for (int rb = 0; rb < 2; ++rb)
            #pragma unroll
            for (int v = 0; v < 8; ++v) {
                redS[rb * 16 + 8 * half + v][wv] = s[rb][v];
                redQ[rb * 16 + 8 * half + v][wv] = q[rb][v];
            }
    }
    __syncthreads();
    if (tid < MR) {
        float ss = 0.f, qq = 0.f;
        for (int w = 0; w < NW; ++w) { ss += redS[tid][w]; qq += redQ[tid][w]; }
        float mu  = ss * (1.0f / D_);
        float var = qq * (1.0f / D_) - mu * mu;
        stats[tid][0] = mu;
        stats[tid][1] = rsqrtf(var + 1e-5f);
    }
    __syncthreads();

    // ---- apply LN(g2,beta2) + gelu, pack bf16 G tile into Abuf
    #pragma unroll
    for (int j = 0; j < 6; ++j) {
        int n = wv * 96 + j * 16 + ln;
        float gam = g2[n], bet = beta2[n];
        #pragma unroll
        for (int rb = 0; rb < 2; ++rb)
            #pragma unroll
            for (int v = 0; v < 8; ++v) {
                int m = rb * 16 + v + 8 * half;
                float x = acc[rb][j][v] + biasA[j];
                x = (x - stats[m][0]) * stats[m][1];
                x = x * gam + bet;
                Abuf[m * ASTR + n] = f2bf(gelu_exact(x));
            }
    }
    __syncthreads();

    // ---- final GEMM: raw = G(32x768) @ w2b(768x768) + b2b
    #pragma unroll
    for (int rb = 0; rb < 2; ++rb)
        #pragma unroll
        for (int j = 0; j < 6; ++j)
            #pragma unroll
            for (int e = 0; e < 8; ++e) acc[rb][j][e] = 0.f;

    for (int kk = 0; kk < D_; kk += 32) {
        v16bf a0 = load_a_frag(Abuf, ASTR, ln,      half, kk);
        v16bf a1 = load_a_frag(Abuf, ASTR, 16 + ln, half, kk);
        #pragma unroll
        for (int j = 0; j < 6; ++j) {
            int n = wv * 96 + j * 16 + ln;
            v16bf bb = load_b_frag(w2bT, n, D_, half, kk);
            acc[0][j] = __builtin_amdgcn_wmma_f32_16x16x32_bf16(
                false, a0, false, bb, (short)0, acc[0][j], false, false);
            acc[1][j] = __builtin_amdgcn_wmma_f32_16x16x32_bf16(
                false, a1, false, bb, (short)0, acc[1][j], false, false);
        }
    }
    #pragma unroll
    for (int j = 0; j < 6; ++j) {
        int n = wv * 96 + j * 16 + ln;
        float bias = b2b[n];
        #pragma unroll
        for (int rb = 0; rb < 2; ++rb)
            #pragma unroll
            for (int v = 0; v < 8; ++v) {
                int m = rb * 16 + v + 8 * half;
                raw_out[(r0 + m) * D_ + n] = acc[rb][j][v] + bias;
            }
    }
}

// ---------------- segment max of raw (fp32) + pos_embed -> out
__global__ void segmax2_kernel(const float* __restrict__ raw, const int* __restrict__ poff,
                               const float* __restrict__ pos_embed, float* __restrict__ out) {
    const int p = blockIdx.x, b = blockIdx.y;
    const int t0 = poff[p], t1 = poff[p + 1];
    for (int d = threadIdx.x; d < D_; d += blockDim.x) {
        float m = -INFINITY;
        for (int t = t0; t < t1; ++t)
            m = fmaxf(m, raw[((long long)b * T_ + t) * D_ + d]);
        out[((long long)b * P_ + p) * D_ + d] = m + pos_embed[p * D_ + d];
    }
}

extern "C" void kernel_launch(void* const* d_in, const int* in_sizes, int n_in,
                              void* d_out, int out_size, void* d_ws, size_t ws_size,
                              hipStream_t stream) {
    const float* latent      = (const float*)d_in[0];
    const int*   patch_index = (const int*)  d_in[1];
    const int*   patch_off   = (const int*)  d_in[2];
    const float* pos_embed   = (const float*)d_in[3];
    const float* w1a  = (const float*)d_in[4];
    const float* b1a  = (const float*)d_in[5];
    const float* w1b  = (const float*)d_in[6];
    const float* b1b  = (const float*)d_in[7];
    const float* w2a  = (const float*)d_in[8];
    const float* b2a  = (const float*)d_in[9];
    const float* g2   = (const float*)d_in[10];
    const float* beta2= (const float*)d_in[11];
    const float* w2b  = (const float*)d_in[12];
    const float* b2b  = (const float*)d_in[13];

    // workspace layout
    char* ws = (char*)d_ws;
    size_t off = 0;
    int* seg = (int*)(ws + off);                          off += (size_t)T_ * 4;            // 128 KB
    unsigned short* w1bT = (unsigned short*)(ws + off);   off += (size_t)D_ * D_ * 2;       // 1.125 MB
    unsigned short* w2aT = (unsigned short*)(ws + off);   off += (size_t)D_ * TWO_D_ * 2;   // 2.25 MB
    unsigned short* w2bT = (unsigned short*)(ws + off);   off += (size_t)D_ * D_ * 2;       // 1.125 MB
    float* yws = (float*)(ws + off);                      off += (size_t)B_ * P_ * D_ * 4;  // 3 MB
    unsigned short* h1ws = (unsigned short*)(ws + off);   off += (size_t)B_ * T_ * D_ * 2;  // 96 MB

    float* outp = (float*)d_out;                       // (B, P, D)
    float* rawp = outp + (size_t)B_ * P_ * D_;         // (B, T, D)

    // 1) weight transpose + bf16 convert
    {
        long long n1 = (long long)D_ * D_;
        convT_kernel<<<(unsigned)((n1 + 255) / 256), 256, 0, stream>>>(w1b, w1bT, D_, D_);
        long long n2 = (long long)D_ * TWO_D_;
        convT_kernel<<<(unsigned)((n2 + 255) / 256), 256, 0, stream>>>(w2a, w2aT, TWO_D_, D_);
        convT_kernel<<<(unsigned)((n1 + 255) / 256), 256, 0, stream>>>(w2b, w2bT, D_, D_);
    }
    // 2) token -> segment map
    seg_kernel<<<T_ / 256, 256, 0, stream>>>(patch_off, seg);

    // 3) stage 1 (fused input GEMM + LN + gelu + WMMA GEMM)
    const int nblk = (B_ * T_) / MR;   // 2048
    stage1_kernel<<<nblk, TPB, 0, stream>>>(latent, patch_index, w1a, b1a, w1bT, b1b, h1ws);

    // 4) per-patch max pooling
    segmax1_kernel<<<dim3(P_, B_), 256, 0, stream>>>(h1ws, patch_off, yws);

    // 5) stage 2 (concat GEMM + LN + gelu + WMMA GEMM) -> raw_latent
    stage2_kernel<<<nblk, TPB, 0, stream>>>(h1ws, yws, seg, w2aT, b2a, g2, beta2,
                                            w2bT, b2b, rawp);

    // 6) final pooling + positional embedding -> out
    segmax2_kernel<<<dim3(P_, B_), 256, 0, stream>>>(rawp, patch_off, pos_embed, outp);
}
